// Block_32401233281211
// MI455X (gfx1250) — compile-verified, matrix operand order checked by reading, not run on hardware
//
#include <hip/hip_runtime.h>
#include <hip/hip_bf16.h>
#include <math.h>
#include <stdint.h>

// ---------------------------------------------------------------------------
// Types for CDNA5 WMMA (wave32)
// ---------------------------------------------------------------------------
typedef _Float16 v16h __attribute__((ext_vector_type(16)));
typedef _Float16 v8h  __attribute__((ext_vector_type(8)));
typedef float    v8f  __attribute__((ext_vector_type(8)));

// Problem constants
#define BATCH    8
#define HH       64
#define WW       64
#define EMBED    768
#define HEADS    12
#define HEAD_DIM 64
#define FF       3072
#define WS       14
#define NWIN     200            // 8 * 5 * 5
#define NTOK     196            // 14*14
#define NROWS    39200          // NWIN*NTOK  (multiple of 32)
#define MPAD     208            // query rows padded to mult of 16
#define KPAD     224            // key rows padded to mult of 32
#define NPIX     32768          // 8*64*64
#define SCALE_F  0.125f         // 64^-0.5
#define EPS_F    1e-6f

// ---------------------------------------------------------------------------
// WMMA fragment loader: per lane two contiguous 16-byte chunks.
// A-matrix 16x32 f16 layout (ISA 7.12.2): lane = (row%16) + 16*h;
// elements 0..7  <- K = 8h .. 8h+7
// elements 8..15 <- K = 16+8h .. 23+8h
// ---------------------------------------------------------------------------
__device__ __forceinline__ v16h load_frag(const _Float16* __restrict__ base,
                                          int ld, int row, int k0, int h) {
  const _Float16* p = base + (long long)row * ld + k0;
  v8h lo = *(const v8h*)(p + 8 * h);
  v8h hi = *(const v8h*)(p + 16 + 8 * h);
  v16h r;
#pragma unroll
  for (int i = 0; i < 8; ++i) { r[i] = lo[i]; r[i + 8] = hi[i]; }
  return r;
}

__device__ __forceinline__ v16h frag_from_lds(const _Float16* p, int h) {
  v8h lo = *(const v8h*)(p + 8 * h);
  v8h hi = *(const v8h*)(p + 16 + 8 * h);
  v16h r;
#pragma unroll
  for (int i = 0; i < 8; ++i) { r[i] = lo[i]; r[i + 8] = hi[i]; }
  return r;
}

__device__ __forceinline__ v8f wmma_f16(v16h a, v16h b, v8f c) {
  return __builtin_amdgcn_wmma_f32_16x16x32_f16(false, a, false, b,
                                                (short)0, c, false, false);
}

// CDNA5 async global->LDS copy (ASYNCcnt) : 16 bytes per lane per issue.
// Low 32 bits of a generic pointer into LDS == LDS byte offset (ISA 10.2).
__device__ __forceinline__ void async_copy_b128(const _Float16* gsrc,
                                                _Float16* lds_dst) {
  unsigned lds = (unsigned)(uintptr_t)lds_dst;
  unsigned long long ga = (unsigned long long)(uintptr_t)gsrc;
  asm volatile("global_load_async_to_lds_b128 %0, %1, off"
               :: "v"(lds), "v"(ga)
               : "memory");
}
__device__ __forceinline__ void wait_async0() {
  asm volatile("s_wait_asynccnt 0" ::: "memory");
}

__device__ __forceinline__ float gelu_exact(float v) {
  return 0.5f * v * (1.0f + erff(v * 0.70710678118654752f));
}

// ---------------------------------------------------------------------------
// Big weight GEMM (LDS-staged, async double-buffered).
// C[M,N] = A[M,K](f16 row-major) * Bt[N,K](f16) + bias + (GELU) + (residual)
// Block = 128 thr = 4 waves arranged 2(M) x 2(N); wave tile = 32 x 64.
// B panel (128 cols x 32 k) is staged per k-step into LDS with
// global_load_async_to_lds_b128 (double-buffered), read as ds_load_b128.
// Requires: M % 32 == 0, K % 32 == 0, N % 128 == 0.
// grid = (N/128, ceil(M/64))
// ---------------------------------------------------------------------------
template <typename OutT, bool GELU_EPI>
__global__ void wmma_gemm_lds(const _Float16* __restrict__ A, int lda,
                              const _Float16* __restrict__ Bt, int ldb,
                              OutT* __restrict__ C, int ldc,
                              const float* __restrict__ bias,
                              const float* __restrict__ res, int ldr,
                              int M, int N, int K) {
  __shared__ _Float16 sB[2][128][32];          // 16 KB, double buffered
  const int tid  = threadIdx.x;
  const int wave = tid >> 5;
  const int lane = tid & 31;
  const int mi = wave >> 1;                    // 0..1 : M sub-block
  const int ni = wave & 1;                     // 0..1 : N sub-block
  const int m0   = blockIdx.y * 64 + mi * 32;
  const int nblk = blockIdx.x * 128;
  const bool mvalid = (m0 < M);                // wave-uniform; no early return
  const int l16 = lane & 15;
  const int h   = lane >> 4;
  const int am0 = mvalid ? (m0 + l16)      : l16;
  const int am1 = mvalid ? (m0 + 16 + l16) : l16;

  // ---- prologue: async-stage B(k=0) panel, preload A(k=0) frags ----
  {
    const _Float16* g = Bt + (long long)(nblk + tid) * ldb;
#pragma unroll
    for (int p2 = 0; p2 < 4; ++p2)
      async_copy_b128(g + p2 * 8, &sB[0][tid][p2 * 8]);
  }
  v16h a0 = load_frag(A, lda, am0, 0, h);
  v16h a1 = load_frag(A, lda, am1, 0, h);
  wait_async0();
  __syncthreads();

  v8f acc[8] = {};                             // [tn] for a0, [4+tn] for a1
  const int nsteps = K / 32;
  for (int s = 0; s < nsteps; ++s) {
    const int cur = s & 1, nxt = cur ^ 1;
    const int k = s * 32;
    v16h a0n = a0, a1n = a1;
    if (s + 1 < nsteps) {
      const _Float16* g = Bt + (long long)(nblk + tid) * ldb + k + 32;
#pragma unroll
      for (int p2 = 0; p2 < 4; ++p2)
        async_copy_b128(g + p2 * 8, &sB[nxt][tid][p2 * 8]);
      a0n = load_frag(A, lda, am0, k + 32, h);
      a1n = load_frag(A, lda, am1, k + 32, h);
    }
#pragma unroll
    for (int t = 0; t < 4; ++t) {
      const int n = ni * 64 + t * 16 + l16;    // row within 128-col B panel
      v16h b = frag_from_lds(&sB[cur][n][0], h);
      acc[t]     = wmma_f16(a0, b, acc[t]);
      acc[4 + t] = wmma_f16(a1, b, acc[4 + t]);
    }
    wait_async0();                             // next B panel resident
    __syncthreads();                           // all waves done with sB[cur]
    a0 = a0n; a1 = a1n;
  }

  if (!mvalid) return;                         // after last barrier: safe
#pragma unroll
  for (int tm = 0; tm < 2; ++tm) {
#pragma unroll
    for (int t = 0; t < 4; ++t) {
      const int n = nblk + ni * 64 + t * 16 + l16;
      const float bv = bias ? bias[n] : 0.0f;
#pragma unroll
      for (int r = 0; r < 8; ++r) {
        const int m = m0 + tm * 16 + r + 8 * h;
        float v = acc[tm * 4 + t][r] + bv;
        if (GELU_EPI) v = gelu_exact(v);
        if (res) v += res[(long long)m * ldr + n];
        C[(long long)m * ldc + n] = (OutT)v;
      }
    }
  }
}

// ---------------------------------------------------------------------------
// Small batched GEMM (attention): direct-from-global fragments.
// Block = 128 thr = 4 waves; wave computes 16 x (16*TN).
// grid = (N/(16*TN), ceil(M/64), batch)
// ---------------------------------------------------------------------------
template <typename OutT, int TN>
__global__ void wmma_gemm_batched(const _Float16* __restrict__ A, int lda,
                                  long long sA,
                                  const _Float16* __restrict__ Bt, int ldb,
                                  long long sB_,
                                  OutT* __restrict__ C, int ldc, long long sC,
                                  int M, int N, int K) {
  const int wave = threadIdx.x >> 5;
  const int lane = threadIdx.x & 31;
  const int m0 = blockIdx.y * 64 + wave * 16;
  if (m0 >= M) return;                         // wave-uniform
  const int n0 = blockIdx.x * (16 * TN);
  const long long bz = blockIdx.z;
  const _Float16* Ab = A + bz * sA;
  const _Float16* Bb = Bt + bz * sB_;
  OutT* Cb = C + bz * sC;
  const int l16 = lane & 15;
  const int h   = lane >> 4;
  const int am  = m0 + l16;

  v8f acc[TN] = {};
  for (int k0 = 0; k0 < K; k0 += 32) {
    v16h a = load_frag(Ab, lda, am, k0, h);
#pragma unroll
    for (int t = 0; t < TN; ++t) {
      v16h b = load_frag(Bb, ldb, n0 + t * 16 + l16, k0, h);
      acc[t] = wmma_f16(a, b, acc[t]);
    }
  }
#pragma unroll
  for (int t = 0; t < TN; ++t) {
    const int n = n0 + t * 16 + l16;
#pragma unroll
    for (int r = 0; r < 8; ++r) {
      const int m = m0 + r + 8 * h;
      Cb[(long long)m * ldc + n] = (OutT)acc[t][r];
    }
  }
}

// ---------------------------------------------------------------------------
// LayerNorm1 fused with window partition -> f16 rows (padded tokens -> 0)
// ---------------------------------------------------------------------------
__global__ void ln1_window(const float* __restrict__ x,
                           const float* __restrict__ g,
                           const float* __restrict__ be,
                           _Float16* __restrict__ Xw) {
  const int wt = blockIdx.x;
  const int w = wt / NTOK, t = wt % NTOK;
  const int b = w / 25, wi = (w % 25) / 5, wj = w % 5;
  const int p = t / WS, qq = t % WS;
  const int hy = wi * WS + p, wxp = wj * WS + qq;
  const int tid = threadIdx.x;
  _Float16* outp = Xw + (long long)wt * EMBED;
  if (hy >= HH || wxp >= WW) {
    for (int c = tid; c < EMBED; c += 256) outp[c] = (_Float16)0.0f;
    return;
  }
  const float* xp = x + (((long long)(b * HH + hy)) * WW + wxp) * EMBED;
  float v0 = xp[tid], v1 = xp[tid + 256], v2 = xp[tid + 512];
  __shared__ float rs[256], rq[256];
  rs[tid] = v0 + v1 + v2;
  rq[tid] = v0 * v0 + v1 * v1 + v2 * v2;
  __syncthreads();
  for (int s = 128; s > 0; s >>= 1) {
    if (tid < s) { rs[tid] += rs[tid + s]; rq[tid] += rq[tid + s]; }
    __syncthreads();
  }
  const float mean = rs[0] * (1.0f / EMBED);
  const float var  = rq[0] * (1.0f / EMBED) - mean * mean;
  const float inv  = rsqrtf(var + EPS_F);
  outp[tid]       = (_Float16)((v0 - mean) * inv * g[tid]       + be[tid]);
  outp[tid + 256] = (_Float16)((v1 - mean) * inv * g[tid + 256] + be[tid + 256]);
  outp[tid + 512] = (_Float16)((v2 - mean) * inv * g[tid + 512] + be[tid + 512]);
}

// LayerNorm2 over residual stream y [NPIX, 768] -> f16
__global__ void ln2_rows(const float* __restrict__ y,
                         const float* __restrict__ g,
                         const float* __restrict__ be,
                         _Float16* __restrict__ h2) {
  const int row = blockIdx.x;
  const int tid = threadIdx.x;
  const float* xp = y + (long long)row * EMBED;
  _Float16* outp = h2 + (long long)row * EMBED;
  float v0 = xp[tid], v1 = xp[tid + 256], v2 = xp[tid + 512];
  __shared__ float rs[256], rq[256];
  rs[tid] = v0 + v1 + v2;
  rq[tid] = v0 * v0 + v1 * v1 + v2 * v2;
  __syncthreads();
  for (int s = 128; s > 0; s >>= 1) {
    if (tid < s) { rs[tid] += rs[tid + s]; rq[tid] += rq[tid + s]; }
    __syncthreads();
  }
  const float mean = rs[0] * (1.0f / EMBED);
  const float var  = rq[0] * (1.0f / EMBED) - mean * mean;
  const float inv  = rsqrtf(var + EPS_F);
  outp[tid]       = (_Float16)((v0 - mean) * inv * g[tid]       + be[tid]);
  outp[tid + 256] = (_Float16)((v1 - mean) * inv * g[tid + 256] + be[tid + 256]);
  outp[tid + 512] = (_Float16)((v2 - mean) * inv * g[tid + 512] + be[tid + 512]);
}

// Transpose + convert weight [K,N] f32 -> [N,K] f16 (B operand storage)
__global__ void transpose_to_f16(const float* __restrict__ in,
                                 _Float16* __restrict__ out, int K, int N) {
  long long idx = (long long)blockIdx.x * 256 + threadIdx.x;
  if (idx >= (long long)K * N) return;
  int k = (int)(idx / N), n = (int)(idx % N);
  out[(long long)n * K + k] = (_Float16)in[idx];
}

// Split qkv f16 [NROWS, 2304] into padded per-(window,head) buffers
__global__ void extract_qkv(const _Float16* __restrict__ qkv,
                            _Float16* __restrict__ q,
                            _Float16* __restrict__ k,
                            _Float16* __restrict__ vT) {
  long long idx = (long long)blockIdx.x * 256 + threadIdx.x;
  if (idx >= (long long)NWIN * HEADS * KPAD * HEAD_DIM) return;
  const int d   = (int)(idx % HEAD_DIM);
  const int r   = (int)((idx / HEAD_DIM) % KPAD);
  const int whh = (int)(idx / (HEAD_DIM * KPAD));
  const int w = whh / HEADS, hh = whh % HEADS;
  const bool valid = (r < NTOK);
  const long long src =
      ((long long)(w * NTOK + r)) * (3 * EMBED) + hh * HEAD_DIM + d;
  _Float16 qv = valid ? qkv[src]             : (_Float16)0.0f;
  _Float16 kv = valid ? qkv[src + EMBED]     : (_Float16)0.0f;
  _Float16 vv = valid ? qkv[src + 2 * EMBED] : (_Float16)0.0f;
  if (r < MPAD) q[((long long)whh * MPAD + r) * HEAD_DIM + d] = qv;
  k[((long long)whh * KPAD + r) * HEAD_DIM + d] = kv;
  vT[((long long)whh * HEAD_DIM + d) * KPAD + r] = vv;
}

// Fused rel-pos bias + masked softmax. One block per (whh, query-row).
__global__ void bias_softmax(const float* __restrict__ S,
                             const _Float16* __restrict__ q,
                             const float* __restrict__ rph,
                             const float* __restrict__ rpw,
                             _Float16* __restrict__ P) {
  const int n   = blockIdx.x;
  const int whh = blockIdx.y;
  const int tid = threadIdx.x;
  __shared__ float relh[WS], relw[WS], qs[HEAD_DIM], red[256];

  const _Float16* qrow = q + ((long long)whh * MPAD + n) * HEAD_DIM;
  if (tid < HEAD_DIM) qs[tid] = (float)qrow[tid];
  __syncthreads();

  const bool rowvalid = (n < NTOK);
  const int p = n / WS, qq = n % WS;
  if (rowvalid) {
    if (tid < WS) {
      const float* R = rph + (p - tid + WS - 1) * HEAD_DIM;
      float s = 0.f;
      for (int d = 0; d < HEAD_DIM; ++d) s += qs[d] * R[d];
      relh[tid] = s;
    } else if (tid >= 32 && tid < 32 + WS) {
      const int kw = tid - 32;
      const float* R = rpw + (qq - kw + WS - 1) * HEAD_DIM;
      float s = 0.f;
      for (int d = 0; d < HEAD_DIM; ++d) s += qs[d] * R[d];
      relw[kw] = s;
    }
  }
  __syncthreads();

  const long long base = ((long long)whh * MPAD + n) * KPAD;
  const int m = tid;
  float sval = -1e30f;
  if (m < NTOK) {
    sval = S[base + m] * SCALE_F;
    if (rowvalid) sval += relh[m / WS] + relw[m % WS];
  }
  red[tid] = sval;
  __syncthreads();
  for (int s = 128; s > 0; s >>= 1) {
    if (tid < s) red[tid] = fmaxf(red[tid], red[tid + s]);
    __syncthreads();
  }
  const float mx = red[0];
  __syncthreads();
  float e = (m < NTOK) ? expf(sval - mx) : 0.0f;
  red[tid] = e;
  __syncthreads();
  for (int s = 128; s > 0; s >>= 1) {
    if (tid < s) red[tid] += red[tid + s];
    __syncthreads();
  }
  const float inv = 1.0f / red[0];
  if (m < KPAD) P[base + m] = (_Float16)((m < NTOK) ? e * inv : 0.0f);
}

// Gather O [2400][MPAD][64] -> proj A matrix f16 [NROWS, 768]
__global__ void merge_heads(const _Float16* __restrict__ O,
                            _Float16* __restrict__ Ap) {
  long long idx = (long long)blockIdx.x * 256 + threadIdx.x;
  if (idx >= (long long)NROWS * EMBED) return;
  const int c = (int)(idx % EMBED);
  const long long row = idx / EMBED;
  const int w = (int)(row / NTOK), t = (int)(row % NTOK);
  const int head = c / HEAD_DIM, d = c % HEAD_DIM;
  Ap[idx] = O[((long long)(w * HEADS + head) * MPAD + t) * HEAD_DIM + d];
}

// Window un-partition + first residual:  y = x + unpart(proj_out)
__global__ void unpart_residual(const float* __restrict__ x,
                                const float* __restrict__ Aout,
                                float* __restrict__ y) {
  long long idx = (long long)blockIdx.x * 256 + threadIdx.x;
  if (idx >= (long long)NPIX * EMBED) return;
  const int c = (int)(idx % EMBED);
  const long long pix = idx / EMBED;
  const int b  = (int)(pix / (HH * WW));
  const int hy = (int)((pix / WW) % HH);
  const int wx = (int)(pix % WW);
  const int wi = hy / WS, p = hy % WS;
  const int wj = wx / WS, qq = wx % WS;
  const int w = b * 25 + wi * 5 + wj;
  const int t = p * WS + qq;
  y[idx] = x[idx] + Aout[((long long)w * NTOK + t) * EMBED + c];
}

// ---------------------------------------------------------------------------
// Host orchestration
// ---------------------------------------------------------------------------
extern "C" void kernel_launch(void* const* d_in, const int* in_sizes, int n_in,
                              void* d_out, int out_size, void* d_ws,
                              size_t ws_size, hipStream_t stream) {
  const float* x      = (const float*)d_in[0];
  const float* g1     = (const float*)d_in[1];
  const float* beta1  = (const float*)d_in[2];
  const float* w_qkv  = (const float*)d_in[3];
  const float* b_qkv  = (const float*)d_in[4];
  const float* w_proj = (const float*)d_in[5];
  const float* b_proj = (const float*)d_in[6];
  const float* rph    = (const float*)d_in[7];
  const float* rpw    = (const float*)d_in[8];
  const float* g2     = (const float*)d_in[9];
  const float* beta2  = (const float*)d_in[10];
  const float* w_fc1  = (const float*)d_in[11];
  const float* b_fc1  = (const float*)d_in[12];
  const float* w_fc2  = (const float*)d_in[13];
  const float* b_fc2  = (const float*)d_in[14];
  float* out = (float*)d_out;

  char* ws = (char*)d_ws;
  size_t off = 0;
  auto carve = [&](size_t bytes) -> char* {
    char* pp = ws + off;
    off += (bytes + 255) & ~size_t(255);
    return pp;
  };
  _Float16* Xw     = (_Float16*)carve((size_t)NROWS * EMBED * 2);
  _Float16* wqkvT  = (_Float16*)carve((size_t)3 * EMBED * EMBED * 2);
  _Float16* wprojT = (_Float16*)carve((size_t)EMBED * EMBED * 2);
  _Float16* w1T    = (_Float16*)carve((size_t)FF * EMBED * 2);
  _Float16* w2T    = (_Float16*)carve((size_t)EMBED * FF * 2);
  _Float16* qkvh   = (_Float16*)carve((size_t)NROWS * 3 * EMBED * 2);
  _Float16* qb     = (_Float16*)carve((size_t)2400 * MPAD * HEAD_DIM * 2);
  _Float16* kb     = (_Float16*)carve((size_t)2400 * KPAD * HEAD_DIM * 2);
  _Float16* vTb    = (_Float16*)carve((size_t)2400 * HEAD_DIM * KPAD * 2);
  float*    Sb     = (float*)   carve((size_t)2400 * MPAD * KPAD * 4);
  _Float16* Pb     = (_Float16*)carve((size_t)2400 * MPAD * KPAD * 2);
  _Float16* Ob     = (_Float16*)carve((size_t)2400 * MPAD * HEAD_DIM * 2);
  _Float16* Aproj  = (_Float16*)carve((size_t)NROWS * EMBED * 2);
  float*    Aout   = (float*)   carve((size_t)NROWS * EMBED * 4);
  float*    yb     = (float*)   carve((size_t)NPIX * EMBED * 4);
  _Float16* h2b    = (_Float16*)carve((size_t)NPIX * EMBED * 2);
  _Float16* Gb     = (_Float16*)carve((size_t)NPIX * FF * 2);

  // 1) LN1 + window partition -> f16
  ln1_window<<<NROWS, 256, 0, stream>>>(x, g1, beta1, Xw);

  // 2) Weight transpose/convert f32 -> f16, N-major
  auto tgrid = [](long long n) { return (unsigned)((n + 255) / 256); };
  transpose_to_f16<<<tgrid((long long)EMBED * 3 * EMBED), 256, 0, stream>>>(
      w_qkv, wqkvT, EMBED, 3 * EMBED);
  transpose_to_f16<<<tgrid((long long)EMBED * EMBED), 256, 0, stream>>>(
      w_proj, wprojT, EMBED, EMBED);
  transpose_to_f16<<<tgrid((long long)EMBED * FF), 256, 0, stream>>>(
      w_fc1, w1T, EMBED, FF);
  transpose_to_f16<<<tgrid((long long)FF * EMBED), 256, 0, stream>>>(
      w_fc2, w2T, FF, EMBED);

  // 3) qkv GEMM (LDS/async): [39200,768] x [768,2304] + b_qkv -> f16
  wmma_gemm_lds<_Float16, false>
      <<<dim3(3 * EMBED / 128, (NROWS + 63) / 64), 128, 0, stream>>>(
          Xw, EMBED, wqkvT, EMBED, qkvh, 3 * EMBED, b_qkv,
          nullptr, 0, NROWS, 3 * EMBED, EMBED);

  // 4) Split + pad q/k/vT
  extract_qkv<<<tgrid((long long)2400 * KPAD * HEAD_DIM), 256, 0, stream>>>(
      qkvh, qb, kb, vTb);

  // 5) Scores: batched [208,64] x [64,224] -> f32 (2400 batches)
  wmma_gemm_batched<float, 2>
      <<<dim3(KPAD / 32, (MPAD + 63) / 64, 2400), 128, 0, stream>>>(
          qb, HEAD_DIM, (long long)MPAD * HEAD_DIM,
          kb, HEAD_DIM, (long long)KPAD * HEAD_DIM,
          Sb, KPAD, (long long)MPAD * KPAD, MPAD, KPAD, HEAD_DIM);

  // 6) rel-pos bias + masked softmax -> f16 P
  bias_softmax<<<dim3(MPAD, 2400), 256, 0, stream>>>(Sb, qb, rph, rpw, Pb);

  // 7) P @ V: batched [208,224] x [224,64] -> f16 O
  wmma_gemm_batched<_Float16, 4>
      <<<dim3(1, (MPAD + 63) / 64, 2400), 128, 0, stream>>>(
          Pb, KPAD, (long long)MPAD * KPAD,
          vTb, KPAD, (long long)HEAD_DIM * KPAD,
          Ob, HEAD_DIM, (long long)MPAD * HEAD_DIM, MPAD, HEAD_DIM, KPAD);

  // 8) merge heads -> proj A matrix
  merge_heads<<<tgrid((long long)NROWS * EMBED), 256, 0, stream>>>(Ob, Aproj);

  // 9) proj GEMM (LDS/async) + bias -> f32
  wmma_gemm_lds<float, false>
      <<<dim3(EMBED / 128, (NROWS + 63) / 64), 128, 0, stream>>>(
          Aproj, EMBED, wprojT, EMBED, Aout, EMBED, b_proj,
          nullptr, 0, NROWS, EMBED, EMBED);

  // 10) window unpartition + residual -> y
  unpart_residual<<<tgrid((long long)NPIX * EMBED), 256, 0, stream>>>(
      x, Aout, yb);

  // 11) LN2 -> f16
  ln2_rows<<<NPIX, 256, 0, stream>>>(yb, g2, beta2, h2b);

  // 12) fc1 GEMM (LDS/async) + bias + exact GELU -> f16
  wmma_gemm_lds<_Float16, true>
      <<<dim3(FF / 128, NPIX / 64), 128, 0, stream>>>(
          h2b, EMBED, w1T, EMBED, Gb, FF, b_fc1,
          nullptr, 0, NPIX, FF, EMBED);

  // 13) fc2 GEMM (LDS/async) + bias + residual(y) -> d_out (f32)
  wmma_gemm_lds<float, false>
      <<<dim3(EMBED / 128, NPIX / 64), 128, 0, stream>>>(
          Gb, FF, w2T, FF, out, EMBED, b_fc2,
          yb, EMBED, NPIX, EMBED, FF);
}
